// ResRnn_24601572671792
// MI455X (gfx1250) — matrix operand order repaired and checked by reading, not compiled, not run on hardware
//
#include <hip/hip_runtime.h>
#include <hip/hip_bf16.h>

// ---------------- problem constants ----------------
constexpr int   INPUT_W = 256;
constexpr int   STATE_W = 1792;
constexpr int   SW      = 2048;          // stream / s width
constexpr int   SEQ     = 512;
constexpr int   BATCH   = 256;
constexpr float LIN     = 0.99999f;
constexpr float ONEML   = 1.0f - 0.99999f;

// ---------------- partitioning ----------------
constexpr int NWG   = 64;                // persistent workgroups, one per WGP
constexpr int NS    = SW / NWG;          // 32 weight rows (output cols) per WG
constexpr int XCOLS = INPUT_W / NWG;     // 4 x-columns refreshed per WG per step
constexpr int NTHR  = 256;               // 8 waves: each computes a 32x32 patch (2x2 tiles)
constexpr int PITCH = 2072;              // LDS row pitch (bf16 elems): 16B-aligned, bank-staggered
constexpr int LDS_BYTES = 2 * NS * PITCH * 2;   // W1 slice + W2 slice = 265216 B

typedef __attribute__((ext_vector_type(16))) __bf16 v16bf;
typedef __attribute__((ext_vector_type(8)))  float  v8f;

union V16 { v16bf v; uint4 q[2]; unsigned short u[16]; };

__device__ __forceinline__ unsigned short f2bf(float f) {
  __hip_bfloat16 h = __float2bfloat16(f);
  return *reinterpret_cast<unsigned short*>(&h);
}

// device-scope sense(generation) barrier across the persistent WGs
__device__ __forceinline__ void grid_barrier(unsigned* bar) {
  __syncthreads();
  if (threadIdx.x == 0) {
    __threadfence();   // writeback this WG's global stores before arriving
    unsigned gen  = __hip_atomic_load(&bar[1], __ATOMIC_RELAXED, __HIP_MEMORY_SCOPE_AGENT);
    unsigned prev = __hip_atomic_fetch_add(&bar[0], 1u, __ATOMIC_ACQ_REL, __HIP_MEMORY_SCOPE_AGENT);
    if (prev == NWG - 1) {
      __hip_atomic_store(&bar[0], 0u, __ATOMIC_RELAXED, __HIP_MEMORY_SCOPE_AGENT);
      __hip_atomic_fetch_add(&bar[1], 1u, __ATOMIC_ACQ_REL, __HIP_MEMORY_SCOPE_AGENT);
    } else {
      while (__hip_atomic_load(&bar[1], __ATOMIC_ACQUIRE, __HIP_MEMORY_SCOPE_AGENT) == gen)
        __builtin_amdgcn_s_sleep(2);
    }
    __threadfence();   // invalidate stale near caches before consuming peers' data
  }
  __syncthreads();
}

// B operand (32x16, K x N) from a precomputed per-lane LDS base; kb in elements.
// Base already folds: nrow*PITCH + (lo?0:8). Offsets stay small -> DS immediates.
__device__ __forceinline__ v16bf ldB(const unsigned short* p, int kb) {
  V16 b;
  b.q[0] = *reinterpret_cast<const uint4*>(p + kb);
  b.q[1] = *reinterpret_cast<const uint4*>(p + kb + 16);
  return b.v;
}

// A operand (16x32, M x K) from a precomputed per-lane global base; kb in elements.
// Base already folds: mrow*SW + (lo?0:8).
__device__ __forceinline__ v16bf ldA(const unsigned short* p, int kb) {
  V16 a;
  a.q[0] = *reinterpret_cast<const uint4*>(p + kb);
  a.q[1] = *reinterpret_cast<const uint4*>(p + kb + 16);
  return a.v;
}

// one K-chunk worth of operands for a 2x2 tile patch
struct Grp { v16bf a0, a1, b0, b1; };

__device__ __forceinline__ Grp load_grp(const unsigned short* pA0, const unsigned short* pA1,
                                        const unsigned short* pB0, const unsigned short* pB1,
                                        int kb) {
  Grp g;
  g.a0 = ldA(pA0, kb);
  g.a1 = ldA(pA1, kb);
  g.b0 = ldB(pB0, kb);
  g.b1 = ldB(pB1, kb);
  return g;
}

__device__ __forceinline__ void fma4(const Grp& s, v8f& c00, v8f& c01, v8f& c10, v8f& c11) {
  c00 = __builtin_amdgcn_wmma_f32_16x16x32_bf16(false, s.a0, false, s.b0, (short)0, c00, false, false);
  c10 = __builtin_amdgcn_wmma_f32_16x16x32_bf16(false, s.a1, false, s.b0, (short)0, c10, false, false);
  c01 = __builtin_amdgcn_wmma_f32_16x16x32_bf16(false, s.a0, false, s.b1, (short)0, c01, false, false);
  c11 = __builtin_amdgcn_wmma_f32_16x16x32_bf16(false, s.a1, false, s.b1, (short)0, c11, false, false);
}

// full 32x32x2048 K-reduction; copy-free double-buffered pipeline (2 chunks ahead)
__device__ __forceinline__ void gemm22(const unsigned short* pA0, const unsigned short* pA1,
                                       const unsigned short* pB0, const unsigned short* pB1,
                                       v8f& c00, v8f& c01, v8f& c10, v8f& c11) {
  v8f z = {0.f, 0.f, 0.f, 0.f, 0.f, 0.f, 0.f, 0.f};
  c00 = z; c01 = z; c10 = z; c11 = z;
  Grp s0 = load_grp(pA0, pA1, pB0, pB1, 0);
  Grp s1 = load_grp(pA0, pA1, pB0, pB1, 32);
  #pragma unroll 1
  for (int kb = 0; kb < SW - 64; kb += 64) {
    fma4(s0, c00, c01, c10, c11);
    s0 = load_grp(pA0, pA1, pB0, pB1, kb + 64);
    fma4(s1, c00, c01, c10, c11);
    s1 = load_grp(pA0, pA1, pB0, pB1, kb + 96);
  }
  fma4(s0, c00, c01, c10, c11);
  fma4(s1, c00, c01, c10, c11);
}

// s_0 = concat([x_0, (1-L)*initial_state broadcast]) in both fp32 and bf16 buffers
__global__ void rnn_init(const float* __restrict__ x,
                         const float* __restrict__ init_state,
                         float* __restrict__ sF,
                         unsigned short* __restrict__ sB,
                         unsigned* __restrict__ bar) {
  int idx = blockIdx.x * blockDim.x + threadIdx.x;   // 524288 threads
  if (idx == 0) { bar[0] = 0u; bar[1] = 0u; }
  int b = idx >> 11;
  int j = idx & 2047;
  float v = (j < INPUT_W) ? x[b * INPUT_W + j] : ONEML * init_state[j - INPUT_W];
  sF[b * SW + j] = v;          // buffer 0 is "cur" at t=0
  sB[b * SW + j] = f2bf(v);
}

__global__ __launch_bounds__(NTHR, 1)
void rnn_persistent(const float* __restrict__ x,
                    const float* __restrict__ W1, const float* __restrict__ b1,
                    const float* __restrict__ W2, const float* __restrict__ b2,
                    float* __restrict__ out,
                    float* __restrict__ sF,
                    unsigned short* __restrict__ sB,
                    unsigned short* __restrict__ hB,
                    unsigned* __restrict__ bar) {
  extern __shared__ unsigned short lds[];
  unsigned short* ldsW1 = lds;
  unsigned short* ldsW2 = lds + NS * PITCH;

  const int g     = blockIdx.x;
  const int tid   = threadIdx.x;
  const int lane  = tid & 31;
  const int wid   = tid >> 5;          // 0..7 -> 32 batch rows each
  const bool lo   = lane < 16;
  const int l15   = lane & 15;
  const int mrow0 = wid * 32 + l15;    // A rows carried by this lane (2 m-tiles)
  const int mrow1 = mrow0 + 16;
  const int jcol0 = g * NS + l15;      // output stream columns (2 n-tiles)
  const int jcol1 = jcol0 + 16;
  const int c0off = lo ? 0 : 8;        // A K-offset for upper half-wave
  const int boff  = lo ? 0 : 8;        // B K-offset for upper half-wave

  // ---- stage this WG's weight slices into LDS (fp32 -> bf16, once) ----
  for (int idx = tid; idx < NS * 2048; idx += NTHR) {
    int n = idx >> 11, k = idx & 2047;
    ldsW1[n * PITCH + k] = f2bf(W1[(size_t)(g * NS + n) * 2048 + k]);
    ldsW2[n * PITCH + k] = f2bf(W2[(size_t)(g * NS + n) * 2048 + k]);
  }
  __syncthreads();

  // per-lane operand base pointers (inner-loop addressing = base + immediate).
  // The n-tile-1 row offset (16*PITCH elems ~ 66KB) exceeds the 16-bit DS
  // immediate field; keep it opaque so the tile-1 base lives in its OWN
  // hoisted VGPR instead of being re-derived with per-iteration v_adds
  // (which also trip the WMMA->VALU WAR hazard nops).
  int t1off = 16 * PITCH;
  asm volatile("" : "+v"(t1off));      // optimizer barrier, value unchanged
  const unsigned short* pW1B0 = ldsW1 + l15 * PITCH + boff;
  const unsigned short* pW1B1 = pW1B0 + t1off;
  const unsigned short* pW2B0 = ldsW2 + l15 * PITCH + boff;
  const unsigned short* pW2B1 = pW2B0 + t1off;
  const int aoff0 = mrow0 * SW + c0off;
  const int aoff1 = mrow1 * SW + c0off;
  const unsigned short* pHA0 = hB + aoff0;
  const unsigned short* pHA1 = hB + aoff1;

  const float b1c0 = b1[jcol0], b1c1 = b1[jcol1];
  const float b2c0 = b2[jcol0], b2c1 = b2[jcol1];
  const bool  live0 = (jcol0 < STATE_W);   // column survives into the next s
  const bool  live1 = (jcol1 < STATE_W);

  #pragma unroll 1
  for (int t = 0; t < SEQ; ++t) {
    const int cur = t & 1, nxt = cur ^ 1;
    const unsigned short* sBc = sB + (size_t)cur * BATCH * SW;
    const float*          sFc = sF + (size_t)cur * BATCH * SW;
    unsigned short*       sBn = sB + (size_t)nxt * BATCH * SW;
    float*                sFn = sF + (size_t)nxt * BATCH * SW;

    // ================= GEMM1: h = relu(s @ W1^T + b1) =================
    v8f c00, c01, c10, c11;
    gemm22(sBc + aoff0, sBc + aoff1, pW1B0, pW1B1, c00, c01, c10, c11);
    #pragma unroll
    for (int r = 0; r < 8; ++r) {
      int r0 = wid * 32 + r + (lo ? 0 : 8);      // rows of m-tile 0 / 1
      int r1 = r0 + 16;
      float h00 = c00[r] + b1c0; h00 = h00 > 0.f ? h00 : 0.f;
      float h01 = c01[r] + b1c1; h01 = h01 > 0.f ? h01 : 0.f;
      float h10 = c10[r] + b1c0; h10 = h10 > 0.f ? h10 : 0.f;
      float h11 = c11[r] + b1c1; h11 = h11 > 0.f ? h11 : 0.f;
      hB[r0 * SW + jcol0] = f2bf(h00);
      hB[r0 * SW + jcol1] = f2bf(h01);
      hB[r1 * SW + jcol0] = f2bf(h10);
      hB[r1 * SW + jcol1] = f2bf(h11);
    }

    grid_barrier(bar);   // all of h visible device-wide

    // ============ GEMM2: out = LIN*s + (1-LIN)*(h @ W2^T + b2) ============
    gemm22(pHA0, pHA1, pW2B0, pW2B1, c00, c01, c10, c11);
    #pragma unroll
    for (int r = 0; r < 8; ++r) {
      int r0 = wid * 32 + r + (lo ? 0 : 8);
      int r1 = r0 + 16;
      float o00 = LIN * sFc[r0 * SW + jcol0] + ONEML * (c00[r] + b2c0);
      float o01 = LIN * sFc[r0 * SW + jcol1] + ONEML * (c01[r] + b2c1);
      float o10 = LIN * sFc[r1 * SW + jcol0] + ONEML * (c10[r] + b2c0);
      float o11 = LIN * sFc[r1 * SW + jcol1] + ONEML * (c11[r] + b2c1);
      if (t == SEQ - 1) {
        // outputs (256x256) then states (256x1792), flat
        if (jcol0 >= STATE_W) out[r0 * INPUT_W + (jcol0 - STATE_W)] = o00;
        else                  out[BATCH * INPUT_W + r0 * STATE_W + jcol0] = o00;
        if (jcol1 >= STATE_W) out[r0 * INPUT_W + (jcol1 - STATE_W)] = o01;
        else                  out[BATCH * INPUT_W + r0 * STATE_W + jcol1] = o01;
        if (jcol0 >= STATE_W) out[r1 * INPUT_W + (jcol0 - STATE_W)] = o10;
        else                  out[BATCH * INPUT_W + r1 * STATE_W + jcol0] = o10;
        if (jcol1 >= STATE_W) out[r1 * INPUT_W + (jcol1 - STATE_W)] = o11;
        else                  out[BATCH * INPUT_W + r1 * STATE_W + jcol1] = o11;
      } else {
        // stream col j feeds s_{t+1} col j+256
        if (live0) {
          sFn[r0 * SW + (jcol0 + INPUT_W)] = o00;
          sBn[r0 * SW + (jcol0 + INPUT_W)] = f2bf(o00);
          sFn[r1 * SW + (jcol0 + INPUT_W)] = o10;
          sBn[r1 * SW + (jcol0 + INPUT_W)] = f2bf(o10);
        }
        if (live1) {
          sFn[r0 * SW + (jcol1 + INPUT_W)] = o01;
          sBn[r0 * SW + (jcol1 + INPUT_W)] = f2bf(o01);
          sFn[r1 * SW + (jcol1 + INPUT_W)] = o11;
          sBn[r1 * SW + (jcol1 + INPUT_W)] = f2bf(o11);
        }
      }
    }

    // refresh the x part of s_{t+1}: each WG converts 4 columns, once per step
    if (t != SEQ - 1) {
      const float* xn = x + (size_t)(t + 1) * BATCH * INPUT_W;
      #pragma unroll
      for (int idx = tid; idx < BATCH * XCOLS; idx += NTHR) {
        int row = idx >> 2;
        int c   = (idx & (XCOLS - 1)) + g * XCOLS;
        float v = xn[row * INPUT_W + c];
        sFn[row * SW + c] = v;
        sBn[row * SW + c] = f2bf(v);
      }
    }

    grid_barrier(bar);   // next s visible; safe to overwrite hB next step
  }
}

extern "C" void kernel_launch(void* const* d_in, const int* in_sizes, int n_in,
                              void* d_out, int out_size, void* d_ws, size_t ws_size,
                              hipStream_t stream) {
  const float* x  = (const float*)d_in[0];
  const float* s0 = (const float*)d_in[1];
  const float* W1 = (const float*)d_in[2];
  const float* b1 = (const float*)d_in[3];
  const float* W2 = (const float*)d_in[4];
  const float* b2 = (const float*)d_in[5];
  float* out = (float*)d_out;

  char* ws = (char*)d_ws;
  float*          sF  = (float*)(ws);                          // 2 x 256x2048 f32  = 4 MB
  unsigned short* sB  = (unsigned short*)(ws + (4 << 20));     // 2 x 256x2048 bf16 = 2 MB
  unsigned short* hB  = (unsigned short*)(ws + (6 << 20));     // 256x2048 bf16     = 1 MB
  unsigned*       bar = (unsigned*)(ws + (7 << 20));           // counter + generation

  (void)in_sizes; (void)n_in; (void)out_size; (void)ws_size;

  // allow >64KB dynamic LDS (weight-stationary slices: ~259 KB of the 320 KB WGP LDS)
  hipFuncSetAttribute(reinterpret_cast<const void*>(&rnn_persistent),
                      hipFuncAttributeMaxDynamicSharedMemorySize, LDS_BYTES);

  rnn_init<<<(BATCH * SW) / 256, 256, 0, stream>>>(x, s0, sF, sB, bar);
  rnn_persistent<<<NWG, NTHR, LDS_BYTES, stream>>>(x, W1, b1, W2, b2, out,
                                                   sF, sB, hB, bar);
}
